// Attention_11682311045352
// MI455X (gfx1250) — matrix operand order repaired and checked by reading, not compiled, not run on hardware
//
#include <hip/hip_runtime.h>
#include <hip/hip_bf16.h>

// ---------------- problem constants ----------------
#define BB      2
#define NSEQ    4096
#define DIMM    1024
#define NHEADS  16
#define DHEAD   64
#define MF      266     // nb_features
#define MPAD    320     // padded feature count (= 5 * 64: no ragged tiles anywhere)
#define MROWS   8192    // BB*NSEQ

typedef __attribute__((ext_vector_type(16))) __bf16 v16bf;
typedef __attribute__((ext_vector_type(8)))  float  v8f;

static __device__ __forceinline__ v8f v8f_zero() {
    v8f v;
#pragma unroll
    for (int i = 0; i < 8; ++i) v[i] = 0.0f;
    return v;
}

static __device__ __forceinline__ v8f wmma_bf16(v16bf a, v16bf b, v8f c) {
    // (neg_a, A, neg_b, B, c_mod, C, reuse_a, reuse_b)
    return __builtin_amdgcn_wmma_f32_16x16x32_bf16(false, a, false, b, (short)0, c, false, false);
}

// A fragment 16(M)x32(K), smem layout [m][k] (k contiguous, row stride ldk)
// 16 contiguous elements (k = 16*(lane>>4) + i) -> 2x ds_load_b128.
static __device__ __forceinline__ v16bf load_a_frag(const __bf16* s, int ldk,
                                                    int mBase, int kBase, int lane) {
    v16bf a;
    const __bf16* p = s + (mBase + (lane & 15)) * ldk + kBase + 16 * (lane >> 4);
#pragma unroll
    for (int i = 0; i < 16; ++i) a[i] = p[i];
    return a;
}

// B fragment 32(K)x16(N), smem layout [n][k] (k contiguous, row stride ldk)
// 16 contiguous elements -> 2x ds_load_b128.
static __device__ __forceinline__ v16bf load_b_frag(const __bf16* s, int ldk,
                                                    int nBase, int kBase, int lane) {
    v16bf b;
    const __bf16* p = s + (nBase + (lane & 15)) * ldk + kBase + 16 * (lane >> 4);
#pragma unroll
    for (int i = 0; i < 16; ++i) b[i] = p[i];
    return b;
}

// ---------------------------------------------------------------------------
// GEMM: C = A[M,1024] * Bw[1024,1024]^T (+bias), bf16 WMMA compute.
// Tiles: 128x128 per WG (256 thr = 8 waves, 4x2 wave grid), BK=32.
// Software-pipelined: next k-chunk global loads overlap current WMMAs.
// ---------------------------------------------------------------------------
template <bool BIAS>
__global__ __launch_bounds__(256) void gemm_nt_kernel(
    const float* __restrict__ A, const float* __restrict__ Bw,
    float* __restrict__ C, const float* __restrict__ bias) {
    constexpr int K = 1024, N = 1024;
    __shared__ __bf16 As[128 * 32];
    __shared__ __bf16 Bs[128 * 32];
    const int t = threadIdx.x, lane = t & 31, w = t >> 5;
    const int wr = w >> 1, wc = w & 1;
    const int mbase = blockIdx.x * 128, nbase = blockIdx.y * 128;

    v8f acc[2][4];
#pragma unroll
    for (int i = 0; i < 2; ++i)
#pragma unroll
        for (int j = 0; j < 4; ++j) acc[i][j] = v8f_zero();

    float aReg[16], bReg[16];
#pragma unroll
    for (int i = 0; i < 16; ++i) {
        int id = t + 256 * i;
        int r = id >> 5, k = id & 31;
        aReg[i] = A[(size_t)(mbase + r) * K + k];
        bReg[i] = Bw[(size_t)(nbase + r) * K + k];
    }

    for (int kb = 0; kb < K; kb += 32) {
#pragma unroll
        for (int i = 0; i < 16; ++i) {
            int id = t + 256 * i;
            As[id] = (__bf16)aReg[i];
            Bs[id] = (__bf16)bReg[i];
        }
        __syncthreads();
        if (kb + 32 < K) {
#pragma unroll
            for (int i = 0; i < 16; ++i) {
                int id = t + 256 * i;
                int r = id >> 5, k = id & 31;
                aReg[i] = A[(size_t)(mbase + r) * K + kb + 32 + k];
                bReg[i] = Bw[(size_t)(nbase + r) * K + kb + 32 + k];
            }
        }
        if (kb + 64 < K) {  // warm L2 two chunks ahead (global_prefetch_b8)
            int r = t >> 1, k = (t & 1) << 4;
            __builtin_prefetch(&A[(size_t)(mbase + r) * K + kb + 64 + k], 0, 1);
            __builtin_prefetch(&Bw[(size_t)(nbase + r) * K + kb + 64 + k], 0, 1);
        }
        v16bf af[2], bf[4];
#pragma unroll
        for (int mt = 0; mt < 2; ++mt) af[mt] = load_a_frag(As, 32, wr * 32 + 16 * mt, 0, lane);
#pragma unroll
        for (int nt = 0; nt < 4; ++nt) bf[nt] = load_b_frag(Bs, 32, wc * 64 + 16 * nt, 0, lane);
#pragma unroll
        for (int mt = 0; mt < 2; ++mt)
#pragma unroll
            for (int nt = 0; nt < 4; ++nt)
                acc[mt][nt] = wmma_bf16(af[mt], bf[nt], acc[mt][nt]);
        __syncthreads();
    }
#pragma unroll
    for (int nt = 0; nt < 4; ++nt) {
        int col = nbase + wc * 64 + 16 * nt + (lane & 15);
        float bv = BIAS ? bias[col] : 0.0f;
#pragma unroll
        for (int mt = 0; mt < 2; ++mt)
#pragma unroll
            for (int r = 0; r < 8; ++r) {
                int row = mbase + wr * 32 + 16 * mt + r + 8 * (lane >> 4);
                C[(size_t)(row * N + col)] = acc[mt][nt][r] + bv;
            }
    }
}

// ---------------------------------------------------------------------------
// Softmax-kernel feature map: dd = (x*dn) proj^T ; out = ratio*(exp(dd-diag-m)+eps)
// MODE 0: query (per-row max) -> qp ; MODE 1: key pass1 (global max via atomicMax)
// MODE 2: key pass2 (use global max) -> kp.  grid (32 bh, 64 rowblocks), 128 thr.
// ---------------------------------------------------------------------------
template <int MODE>
__global__ __launch_bounds__(128) void dd_kernel(
    const float* __restrict__ QK, const float* __restrict__ proj,
    __bf16* __restrict__ outp, unsigned* __restrict__ kmax) {
    __shared__ __bf16 projs[MPAD * 64];   // 40,960 B
    __shared__ __bf16 qs[64 * 64];        //  8,192 B
    __shared__ float  wmaxLds[4];

    const int t = threadIdx.x, lane = t & 31, w = t >> 5;
    const int bh = blockIdx.x, bb = bh >> 4, hh = bh & 15;
    const int nbase = blockIdx.y * 64;
    const float* src = QK + (size_t)bb * NSEQ * DIMM + hh * DHEAD;

    const float DN = 0.3535533905932738f;  // 64^-0.25

#pragma unroll 8
    for (int i = 0; i < (MPAD * 64) / 128; ++i) {
        int id = t + 128 * i;
        int m = id >> 6, d = id & 63;
        projs[id] = (__bf16)((m < MF) ? proj[m * 64 + d] : 0.0f);
    }
#pragma unroll
    for (int i = 0; i < 32; ++i) {
        int id = t + 128 * i;
        int r = id >> 6, d = id & 63;
        qs[id] = (__bf16)(src[(size_t)(nbase + r) * DIMM + d] * DN);
    }
    // diag per row (lane j of wave w holds row 16w+j); 0.5*dn^2 = 1/16
    float diagv;
    {
        const float* p = src + (size_t)(nbase + 16 * w + (lane & 15)) * DIMM;
        float s = 0.0f;
#pragma unroll 8
        for (int d = 0; d < 64; ++d) { float x = p[d]; s += x * x; }
        diagv = 0.0625f * s;
    }
    __syncthreads();

    const int m0 = 16 * w;
    v16bf a0 = load_a_frag(qs, 64, m0, 0, lane);
    v16bf a1 = load_a_frag(qs, 64, m0, 32, lane);
    const int colBase = lane & 15;

    float rmax[8];
#pragma unroll
    for (int r = 0; r < 8; ++r) rmax[r] = -3.0e38f;

    if (MODE == 0 || MODE == 1) {
        for (int tl = 0; tl < MPAD / 16; ++tl) {
            v8f c = v8f_zero();
            c = wmma_bf16(a0, load_b_frag(projs, 64, 16 * tl, 0, lane), c);
            c = wmma_bf16(a1, load_b_frag(projs, 64, 16 * tl, 32, lane), c);
            bool valid = (16 * tl + colBase) < MF;
#pragma unroll
            for (int r = 0; r < 8; ++r)
                rmax[r] = fmaxf(rmax[r], valid ? c[r] : -3.0e38f);
        }
        // reduce across the 16-lane half (rows live entirely within a half)
#pragma unroll
        for (int r = 0; r < 8; ++r)
#pragma unroll
            for (int s = 1; s < 16; s <<= 1)
                rmax[r] = fmaxf(rmax[r], __shfl_xor(rmax[r], s, 32));
    }

    if (MODE == 1) {
        float vmax = rmax[0];
#pragma unroll
        for (int r = 1; r < 8; ++r) vmax = fmaxf(vmax, rmax[r]);
        vmax = fmaxf(vmax, __shfl_xor(vmax, 16, 32));
        if (lane == 0) wmaxLds[w] = vmax;
        __syncthreads();
        if (t == 0) {
            float m = fmaxf(fmaxf(wmaxLds[0], wmaxLds[1]), fmaxf(wmaxLds[2], wmaxLds[3]));
            unsigned bits = __float_as_uint(m);
            unsigned key = (bits & 0x80000000u) ? ~bits : (bits | 0x80000000u);
            atomicMax(kmax + bh, key);
        }
        return;
    }

    float mrow[8];
    if (MODE == 0) {
#pragma unroll
        for (int r = 0; r < 8; ++r) mrow[r] = rmax[r];
    } else {
        unsigned key = kmax[bh];
        unsigned bits = (key & 0x80000000u) ? (key & 0x7FFFFFFFu) : ~key;
        float gm = __uint_as_float(bits);
#pragma unroll
        for (int r = 0; r < 8; ++r) mrow[r] = gm;
    }
    float drow[8];
#pragma unroll
    for (int r = 0; r < 8; ++r) drow[r] = __shfl(diagv, 8 * (lane >> 4) + r, 32);

    const float RATIO = 0.06131393394849658f;  // 266^-0.5
    const float EPSV  = 1e-4f;
    for (int tl = 0; tl < MPAD / 16; ++tl) {
        v8f c = v8f_zero();
        c = wmma_bf16(a0, load_b_frag(projs, 64, 16 * tl, 0, lane), c);
        c = wmma_bf16(a1, load_b_frag(projs, 64, 16 * tl, 32, lane), c);
        int col = 16 * tl + colBase;
        bool valid = col < MF;
#pragma unroll
        for (int r = 0; r < 8; ++r) {
            int row = nbase + m0 + r + 8 * (lane >> 4);
            float v = valid ? (RATIO * (__expf(c[r] - drow[r] - mrow[r]) + EPSV)) : 0.0f;
            outp[((size_t)bh * NSEQ + row) * MPAD + col] = (__bf16)v;
        }
    }
}

// k_cumsum[bh][m] = sum_n kp[bh][n][m]   (deterministic column sums)
__global__ __launch_bounds__(MPAD) void kcs_kernel(const __bf16* __restrict__ kp,
                                                   float* __restrict__ kcs) {
    int bh = blockIdx.x, t = threadIdx.x;
    const __bf16* p = kp + (size_t)bh * NSEQ * MPAD + t;
    float s = 0.0f;
    for (int n = 0; n < NSEQ; ++n) s += (float)p[(size_t)n * MPAD];
    kcs[bh * MPAD + t] = s;
}

// D_inv[row] = 1 / dot(qp[row,:], kcs[bh,:]) ; one wave per row
__global__ __launch_bounds__(256) void dinv_kernel(const __bf16* __restrict__ qp,
                                                   const float* __restrict__ kcs,
                                                   float* __restrict__ Dinv) {
    int t = threadIdx.x, lane = t & 31, w = t >> 5;
    size_t row = (size_t)blockIdx.x * 8 + w;
    int bh = (int)(row >> 12);
    const __bf16* p = qp + row * MPAD;
    const float* kc = kcs + bh * MPAD;
    float s = 0.0f;
#pragma unroll
    for (int i = 0; i < MPAD / 32; ++i) {
        int m = lane + 32 * i;
        s += (float)p[m] * kc[m];
    }
#pragma unroll
    for (int sh = 16; sh >= 1; sh >>= 1) s += __shfl_xor(s, sh, 32);
    if (lane == 0) Dinv[row] = 1.0f / s;
}

// context[bh][m][e] = sum_n kp[n][m] * v[n][e]   (TN gemm, K = NSEQ)
// LDS staged pre-transposed so all fragment reads are contiguous (ds_load_b128).
// MPAD = 320 = 5 * 64: every m-block is full, no guards anywhere.
__global__ __launch_bounds__(128) void ctx_kernel(const __bf16* __restrict__ kp,
                                                  const float* __restrict__ V,
                                                  float* __restrict__ ctx) {
    __shared__ __bf16 As[64 * 32];  // [m-local][n]  (n contiguous)
    __shared__ __bf16 Bs[64 * 32];  // [e][n]        (n contiguous)
    const int t = threadIdx.x, lane = t & 31, w = t >> 5;
    const int bh = blockIdx.x, mb = blockIdx.y;
    const int bb = bh >> 4, hh = bh & 15;
    const int mbase = mb * 64;
    const float* vsrc = V + (size_t)bb * NSEQ * DIMM + hh * DHEAD;
    const __bf16* ksrc = kp + (size_t)bh * NSEQ * MPAD + mbase;

    v8f acc[4];
#pragma unroll
    for (int i = 0; i < 4; ++i) acc[i] = v8f_zero();

    for (int nb = 0; nb < NSEQ; nb += 32) {
        // A: kp^T tile [64m x 32n]; global read coalesced in m, transpose on LDS write
#pragma unroll
        for (int i = 0; i < 16; ++i) {
            int id = t + 128 * i;
            int m = id & 63, n = id >> 6;
            As[m * 32 + n] = ksrc[(size_t)(nb + n) * MPAD + m];
        }
        // B: v tile [64e x 32n]; global read coalesced in e, transpose on LDS write
#pragma unroll
        for (int i = 0; i < 16; ++i) {
            int id = t + 128 * i;
            int e = id & 63, n = id >> 6;
            Bs[e * 32 + n] = (__bf16)vsrc[(size_t)(nb + n) * DIMM + e];
        }
        __syncthreads();
        v16bf a = load_a_frag(As, 32, 16 * w, 0, lane);
#pragma unroll
        for (int et = 0; et < 4; ++et)
            acc[et] = wmma_bf16(a, load_b_frag(Bs, 32, 16 * et, 0, lane), acc[et]);
        __syncthreads();
    }
#pragma unroll
    for (int et = 0; et < 4; ++et)
#pragma unroll
        for (int r = 0; r < 8; ++r) {
            int m = mbase + 16 * w + r + 8 * (lane >> 4);
            ctx[((size_t)bh * MPAD + m) * 64 + 16 * et + (lane & 15)] = acc[et][r];
        }
}

// O[b, n, h*64+e] = D_inv[n] * sum_m qp[n,m] * ctx[m,e]    (K = MPAD)
__global__ __launch_bounds__(128) void outhead_kernel(const __bf16* __restrict__ qp,
                                                      const float* __restrict__ ctx,
                                                      const float* __restrict__ Dinv,
                                                      float* __restrict__ O) {
    __shared__ __bf16 cs[64 * MPAD];  // [e][m] (m contiguous)  40,960 B
    __shared__ __bf16 qs[64 * 32];    // [n][m-chunk]            4,096 B
    const int t = threadIdx.x, lane = t & 31, w = t >> 5;
    const int bh = blockIdx.x, nbase = blockIdx.y * 64;
    const int bb = bh >> 4, hh = bh & 15;

    // stage ctx transposed: global read coalesced in e, LDS write transposed
    const float* csrc = ctx + (size_t)bh * MPAD * 64;
#pragma unroll 8
    for (int i = 0; i < (64 * MPAD) / 128; ++i) {
        int id = t + 128 * i;
        int e = id & 63, m = id >> 6;
        cs[e * MPAD + m] = (__bf16)csrc[m * 64 + e];
    }

    v8f acc[4];
#pragma unroll
    for (int i = 0; i < 4; ++i) acc[i] = v8f_zero();

    const __bf16* qsrc = qp + ((size_t)bh * NSEQ + nbase) * MPAD;
    for (int mc = 0; mc < MPAD; mc += 32) {
        __syncthreads();
#pragma unroll
        for (int i = 0; i < 16; ++i) {
            int id = t + 128 * i;
            int r = id >> 5, k = id & 31;
            qs[id] = qsrc[(size_t)r * MPAD + mc + k];
        }
        __syncthreads();
        v16bf a = load_a_frag(qs, 32, 16 * w, 0, lane);
#pragma unroll
        for (int et = 0; et < 4; ++et)
            acc[et] = wmma_bf16(a, load_b_frag(cs, MPAD, 16 * et, mc, lane), acc[et]);
    }
#pragma unroll
    for (int r = 0; r < 8; ++r) {
        int nrow = nbase + 16 * w + r + 8 * (lane >> 4);
        float di = Dinv[(size_t)bh * NSEQ + nrow];
#pragma unroll
        for (int et = 0; et < 4; ++et)
            O[((size_t)bb * NSEQ + nrow) * DIMM + hh * DHEAD + 16 * et + (lane & 15)] =
                acc[et][r] * di;
    }
}

__global__ void init_kernel(unsigned* kmax) {
    if (threadIdx.x < 32) kmax[threadIdx.x] = 0u;  // ordered-key for -inf
}

// ---------------------------------------------------------------------------
extern "C" void kernel_launch(void* const* d_in, const int* in_sizes, int n_in,
                              void* d_out, int out_size, void* d_ws, size_t ws_size,
                              hipStream_t stream) {
    const float* x    = (const float*)d_in[0];
    const float* Wq   = (const float*)d_in[1];
    const float* Wk   = (const float*)d_in[2];
    const float* Wv   = (const float*)d_in[3];
    const float* Wo   = (const float*)d_in[4];
    const float* bo   = (const float*)d_in[5];
    const float* proj = (const float*)d_in[6];
    float* out = (float*)d_out;

    char* ws = (char*)d_ws;
    const size_t SZ_QKV = (size_t)MROWS * DIMM * sizeof(float);        // 33.5 MB
    const size_t SZ_P   = (size_t)32 * NSEQ * MPAD * sizeof(__bf16);   // 83.9 MB
    float*    Q    = (float*)(ws);
    float*    K    = (float*)(ws + SZ_QKV);
    float*    V    = (float*)(ws + 2 * SZ_QKV);
    __bf16*   qp   = (__bf16*)(ws + 3 * SZ_QKV);
    __bf16*   kp   = (__bf16*)(ws + 3 * SZ_QKV + SZ_P);
    float*    kcs  = (float*)(ws + 3 * SZ_QKV + 2 * SZ_P);
    unsigned* kmax = (unsigned*)(ws + 3 * SZ_QKV + 2 * SZ_P + 32 * MPAD * sizeof(float));
    float*    ctx  = (float*)((char*)kmax + 256);
    float*    Dinv = (float*)((char*)ctx + (size_t)32 * MPAD * 64 * sizeof(float));
    float*    O    = Q;  // reuse Q buffer after qp is computed

    dim3 gGemm(MROWS / 128, DIMM / 128);
    dim3 gDD(32, NSEQ / 64);
    dim3 gCtx(32, MPAD / 64);

    init_kernel<<<1, 64, 0, stream>>>(kmax);
    gemm_nt_kernel<false><<<gGemm, 256, 0, stream>>>(x, Wq, Q, nullptr);
    gemm_nt_kernel<false><<<gGemm, 256, 0, stream>>>(x, Wk, K, nullptr);
    gemm_nt_kernel<false><<<gGemm, 256, 0, stream>>>(x, Wv, V, nullptr);
    dd_kernel<0><<<gDD, 128, 0, stream>>>(Q, proj, qp, kmax);   // qp
    dd_kernel<1><<<gDD, 128, 0, stream>>>(K, proj, kp, kmax);   // global max of k features
    dd_kernel<2><<<gDD, 128, 0, stream>>>(K, proj, kp, kmax);   // kp
    kcs_kernel<<<32, MPAD, 0, stream>>>(kp, kcs);
    ctx_kernel<<<gCtx, 128, 0, stream>>>(kp, V, ctx);
    dinv_kernel<<<(MROWS * NHEADS) / 8, 256, 0, stream>>>(qp, kcs, Dinv);
    outhead_kernel<<<gDD, 128, 0, stream>>>(qp, ctx, Dinv, O);
    gemm_nt_kernel<true><<<gGemm, 256, 0, stream>>>(O, Wo, out, bo);
}
